// mhvqvae_9998683865097
// MI455X (gfx1250) — matrix operand order repaired and checked by reading, not compiled, not run on hardware
//
#include <hip/hip_runtime.h>
#include <hip/hip_bf16.h>

// ---------------- problem constants (from reference) ----------------
#define N_PTS       16384
#define NUM_EMB     8192
#define D_DIM       512
#define ROWS_PER_BLK 32               // 2 M-tiles of 16 rows per wave
#define COLS_PER_CHUNK 128            // 8 waves * 16 cols
#define NUM_CHUNKS  (NUM_EMB / COLS_PER_CHUNK)   // 64
#define K_CHUNKS    (D_DIM / 32)                  // 16 WMMA K-steps

// LDS padding to spread rows across the 64 banks
#define Z_STRIDE    520               // bf16 elems; 1040 B/row -> +4 banks/row, 16B aligned
#define DIST_STRIDE 132               // f32 elems;  528 B/row -> +4 banks/row

typedef __bf16 bf16;
typedef __attribute__((ext_vector_type(16))) __bf16 v16bf;
typedef __attribute__((ext_vector_type(8)))  __bf16 v8bf;
typedef __attribute__((ext_vector_type(4)))  __bf16 v4bf;
typedef __attribute__((ext_vector_type(8)))  float  v8f;

#define FLT_BIG 3.402823466e+38f

// out layout (flat, return order): z_q_ste [N*D] | vq_loss [1] | k_hot [N*NUM_EMB]
#define LOSS_OFF  ((size_t)N_PTS * D_DIM)
#define KHOT_OFF  (LOSS_OFF + 1)

// ---- dynamic LDS layout (bytes) ----
#define SM_ZHI    0
#define SM_ZLO    (SM_ZHI + ROWS_PER_BLK * Z_STRIDE * 2)            // 33280
#define SM_DIST   (SM_ZLO + ROWS_PER_BLK * Z_STRIDE * 2)            // 66560
#define SM_MD     (SM_DIST + ROWS_PER_BLK * DIST_STRIDE * 4)        // 83456
#define SM_MI     (SM_MD + ROWS_PER_BLK * 32 * 4)                   // 87552
#define SM_RED    (SM_MI + ROWS_PER_BLK * 32 * 4)                   // 91648
#define SM_FIDX   (SM_RED + 256 * 4)                                // 92672
#define SM_TOTAL  (SM_FIDX + ROWS_PER_BLK * 4 * 4)                  // 93184

// ---------------- kernel 1: codebook -> bf16, row norms ----------------
__global__ void prep_codebook(const float* __restrict__ cb,
                              bf16* __restrict__ cbb,
                              float* __restrict__ wnorm) {
    __shared__ float red[128];
    const int row = blockIdx.x;       // one codebook entry per block
    const int t   = threadIdx.x;      // 128 threads, 4 floats each
    const float4 v = ((const float4*)(cb + (size_t)row * D_DIM))[t];
    float s = v.x * v.x + v.y * v.y + v.z * v.z + v.w * v.w;
    v4bf b;
    b[0] = (bf16)v.x; b[1] = (bf16)v.y; b[2] = (bf16)v.z; b[3] = (bf16)v.w;
    ((v4bf*)(cbb + (size_t)row * D_DIM))[t] = b;
    red[t] = s;
    __syncthreads();
    #pragma unroll
    for (int off = 64; off > 0; off >>= 1) {
        if (t < off) red[t] += red[t + off];
        __syncthreads();
    }
    if (t == 0) wnorm[row] = red[0];
}

// ---------------- kernel 2: zero loss + k_hot region ----------------
__global__ void zero_tail(float* __restrict__ out) {
    // LOSS_OFF is a multiple of 4 -> 16B aligned base
    float4* p = (float4*)(out + LOSS_OFF);
    const size_t nvec = ((size_t)N_PTS * NUM_EMB) / 4;  // covers loss + all but last khot elem
    const size_t stride = (size_t)gridDim.x * blockDim.x;
    size_t i = (size_t)blockIdx.x * blockDim.x + threadIdx.x;
    const float4 z4 = make_float4(0.f, 0.f, 0.f, 0.f);
    for (; i < nvec; i += stride) p[i] = z4;
    if (blockIdx.x == 0 && threadIdx.x == 0)
        out[LOSS_OFF + nvec * 4] = 0.0f;                // final scalar of k_hot
}

// ---------------- top-4 (smallest) insertion ----------------
__device__ __forceinline__ void topk_insert(float dd, int idx,
        float& d0, int& i0, float& d1, int& i1,
        float& d2, int& i2, float& d3, int& i3) {
    if (dd < d3) {
        if (dd < d2) {
            d3 = d2; i3 = i2;
            if (dd < d1) {
                d2 = d1; i2 = i1;
                if (dd < d0) { d1 = d0; i1 = i0; d0 = dd; i0 = idx; }
                else         { d1 = dd; i1 = idx; }
            } else { d2 = dd; i2 = idx; }
        } else { d3 = dd; i3 = idx; }
    }
}

__device__ __forceinline__ v16bf cat8(v8bf a, v8bf b) {
    return __builtin_shufflevector(a, b, 0,1,2,3,4,5,6,7,8,9,10,11,12,13,14,15);
}

// ---------------- kernel 3: WMMA distances + top-k + outputs ----------------
__global__ __launch_bounds__(256)
void vq_main(const float* __restrict__ z,
             const float* __restrict__ cb,
             const bf16*  __restrict__ cbb,
             const float* __restrict__ wnorm,
             float* __restrict__ out) {
    extern __shared__ char smem[];
    bf16*  zhi     = (bf16*) (smem + SM_ZHI);
    bf16*  zlo     = (bf16*) (smem + SM_ZLO);
    float* ldsDist = (float*)(smem + SM_DIST);
    float* mD      = (float*)(smem + SM_MD);
    int*   mI      = (int*)  (smem + SM_MI);
    float* red     = (float*)(smem + SM_RED);
    int*   fIdx    = (int*)  (smem + SM_FIDX);

    const int t    = threadIdx.x;
    const int lane = t & 31;
    const int wv   = t >> 5;          // wave id 0..7
    const int r0   = blockIdx.x * ROWS_PER_BLK;
    const int laneM = lane & 15;      // A-row / B-col within 16x16 tile
    const int khalf = lane >> 4;      // K half select per CDNA5 bf16 layouts

    // ---- Phase A: stage z rows as bf16 hi/lo split in LDS ----
    for (int i = t; i < ROWS_PER_BLK * D_DIM; i += 256) {
        const int row = i >> 9, d = i & (D_DIM - 1);
        const float v = z[(size_t)(r0 + row) * D_DIM + d];
        const bf16 h = (bf16)v;
        zhi[row * Z_STRIDE + d] = h;
        zlo[row * Z_STRIDE + d] = (bf16)(v - (float)h);
    }
    __syncthreads();

    // A-row base pointers for the two M-tiles of this lane
    const bf16* zh0 = zhi + (size_t)laneM        * Z_STRIDE;
    const bf16* zh1 = zhi + (size_t)(16 + laneM) * Z_STRIDE;
    const bf16* zl0 = zlo + (size_t)laneM        * Z_STRIDE;
    const bf16* zl1 = zlo + (size_t)(16 + laneM) * Z_STRIDE;

    // private running top-4 (this thread owns row t>>3, cols (t&7) + 8j)
    float d0 = FLT_BIG, d1 = FLT_BIG, d2 = FLT_BIG, d3 = FLT_BIG;
    int   i0 = 0, i1 = 0, i2 = 0, i3 = 0;

    for (int cc = 0; cc < NUM_CHUNKS; ++cc) {
        const int colbase = cc * COLS_PER_CHUNK;
        const int col = colbase + wv * 16 + laneM;   // this lane's B column
        const bf16* bptr = cbb + (size_t)col * D_DIM + khalf * 8;

        if (cc + 1 < NUM_CHUNKS)   // warm next chunk's B tiles
            __builtin_prefetch(cbb + (size_t)(col + COLS_PER_CHUNK) * D_DIM, 0, 3);

        // 4 independent accumulator chains: {tile0,tile1} x {hi,lo}
        v8f aH0 = {0.f,0.f,0.f,0.f,0.f,0.f,0.f,0.f};
        v8f aL0 = {0.f,0.f,0.f,0.f,0.f,0.f,0.f,0.f};
        v8f aH1 = {0.f,0.f,0.f,0.f,0.f,0.f,0.f,0.f};
        v8f aL1 = {0.f,0.f,0.f,0.f,0.f,0.f,0.f,0.f};

        #pragma unroll
        for (int kc = 0; kc < K_CHUNKS; ++kc) {
            const int kb = kc * 32;
            const int o0 = kb + khalf * 8;        // A K = kb + khalf*8  + [0..7]
            const int o1 = kb + 16 + khalf * 8;   //     kb + 16 + khalf*8 + [0..7]
            // B fragment (32x16 bf16): lane holds col, K = kb + khalf*8 + [0..7], kb + 16 + khalf*8 + [0..7]
            const v16bf bb = cat8(*(const v8bf*)(bptr + kb),
                                  *(const v8bf*)(bptr + kb + 16));
            // A fragments (16x32 bf16) for both M-tiles, hi & lo halves
            const v16bf ahi0 = cat8(*(const v8bf*)(zh0 + o0), *(const v8bf*)(zh0 + o1));
            const v16bf alo0 = cat8(*(const v8bf*)(zl0 + o0), *(const v8bf*)(zl0 + o1));
            const v16bf ahi1 = cat8(*(const v8bf*)(zh1 + o0), *(const v8bf*)(zh1 + o1));
            const v16bf alo1 = cat8(*(const v8bf*)(zl1 + o0), *(const v8bf*)(zl1 + o1));
            // compensated split: z.w ~= z_hi.w + z_lo.w, fp32 accumulate,
            // one B fragment feeds 4 WMMAs (doubles intensity per B byte)
            aH0 = __builtin_amdgcn_wmma_f32_16x16x32_bf16(false, ahi0, false, bb, (short)0, aH0, false, false);
            aL0 = __builtin_amdgcn_wmma_f32_16x16x32_bf16(false, alo0, false, bb, (short)0, aL0, false, false);
            aH1 = __builtin_amdgcn_wmma_f32_16x16x32_bf16(false, ahi1, false, bb, (short)0, aH1, false, false);
            aL1 = __builtin_amdgcn_wmma_f32_16x16x32_bf16(false, alo1, false, bb, (short)0, aL1, false, false);
        }

        // distance (minus row-constant ||z||^2): ||w||^2 - 2 z.w
        const float wn = wnorm[col];
        const int rowOff = khalf * 8;    // C/D layout: lanes 16-31 hold M=8..15
        #pragma unroll
        for (int v = 0; v < 8; ++v) {
            ldsDist[(v + rowOff)      * DIST_STRIDE + wv * 16 + laneM] = wn - 2.0f * (aH0[v] + aL0[v]);
            ldsDist[(16 + v + rowOff) * DIST_STRIDE + wv * 16 + laneM] = wn - 2.0f * (aH1[v] + aL1[v]);
        }
        __syncthreads();

        // ---- per-row top-4 scan: 8 threads per row, 16 cols each ----
        {
            const int row = t >> 3, sub = t & 7;
            #pragma unroll
            for (int j = 0; j < 16; ++j) {
                const int cl = sub + j * 8;
                topk_insert(ldsDist[row * DIST_STRIDE + cl], colbase + cl,
                            d0, i0, d1, i1, d2, i2, d3, i3);
            }
        }
        __syncthreads();
    }

    // ---- merge 8 partial top-4s per row ----
    {
        const int row = t >> 3, sub = t & 7;
        mD[row * 32 + sub * 4 + 0] = d0; mI[row * 32 + sub * 4 + 0] = i0;
        mD[row * 32 + sub * 4 + 1] = d1; mI[row * 32 + sub * 4 + 1] = i1;
        mD[row * 32 + sub * 4 + 2] = d2; mI[row * 32 + sub * 4 + 2] = i2;
        mD[row * 32 + sub * 4 + 3] = d3; mI[row * 32 + sub * 4 + 3] = i3;
    }
    __syncthreads();
    if (t < ROWS_PER_BLK) {
        float e0 = FLT_BIG, e1 = FLT_BIG, e2 = FLT_BIG, e3 = FLT_BIG;
        int   j0 = 0, j1 = 0, j2 = 0, j3 = 0;
        for (int c = 0; c < 32; ++c)
            topk_insert(mD[t * 32 + c], mI[t * 32 + c], e0, j0, e1, j1, e2, j2, e3, j3);
        fIdx[t * 4 + 0] = j0; fIdx[t * 4 + 1] = j1;
        fIdx[t * 4 + 2] = j2; fIdx[t * 4 + 3] = j3;
    }
    __syncthreads();

    // ---- k_hot ones ----
    if (t < ROWS_PER_BLK * 4) {
        const int row = t >> 2, j = t & 3;
        out[KHOT_OFF + (size_t)(r0 + row) * NUM_EMB + fIdx[row * 4 + j]] = 1.0f;
    }

    // ---- z_q (== z_q_ste forward) + loss partial ----
    float lossAcc = 0.0f;
    for (int i = t; i < ROWS_PER_BLK * D_DIM; i += 256) {
        const int row = i >> 9, d = i & (D_DIM - 1);
        const int a = fIdx[row * 4 + 0], b = fIdx[row * 4 + 1];
        const int c = fIdx[row * 4 + 2], e = fIdx[row * 4 + 3];
        const float q = 0.25f * (cb[(size_t)a * D_DIM + d] + cb[(size_t)b * D_DIM + d]
                               + cb[(size_t)c * D_DIM + d] + cb[(size_t)e * D_DIM + d]);
        const size_t gi = (size_t)(r0 + row) * D_DIM + d;
        out[gi] = q;
        const float diff = q - z[gi];
        lossAcc += diff * diff;
    }
    red[t] = lossAcc;
    __syncthreads();
    #pragma unroll
    for (int off = 128; off > 0; off >>= 1) {
        if (t < off) red[t] += red[t + off];
        __syncthreads();
    }
    if (t == 0)  // vq_loss = (1 + COMMIT) * mean((z_q - z_e)^2)
        atomicAdd(out + LOSS_OFF, red[0] * (1.25f / ((float)N_PTS * (float)D_DIM)));
}

// ---------------- host launcher ----------------
extern "C" void kernel_launch(void* const* d_in, const int* in_sizes, int n_in,
                              void* d_out, int out_size, void* d_ws, size_t ws_size,
                              hipStream_t stream) {
    (void)in_sizes; (void)n_in; (void)out_size; (void)ws_size;
    const float* z  = (const float*)d_in[0];   // [N, D] fp32
    const float* cb = (const float*)d_in[1];   // [NUM_EMB, D] fp32
    // d_in[2] is k (== 4, hardcoded)
    float* out = (float*)d_out;

    bf16*  cbb   = (bf16*)d_ws;                                          // 8 MB
    float* wnorm = (float*)((char*)d_ws + (size_t)NUM_EMB * D_DIM * sizeof(bf16));

    prep_codebook<<<NUM_EMB, 128, 0, stream>>>(cb, cbb, wnorm);
    zero_tail<<<2048, 256, 0, stream>>>(out);
    vq_main<<<N_PTS / ROWS_PER_BLK, 256, SM_TOTAL, stream>>>(z, cb, cbb, wnorm, out);
}
